// Communication_62534723829954
// MI455X (gfx1250) — matrix operand order repaired and checked by reading, not compiled, not run on hardware
//
#include <hip/hip_runtime.h>
#include <hip/hip_bf16.h>
#include <math.h>

// ---------------------------------------------------------------------------
// Communication module, MI455X (gfx1250, wave32, WMMA).
// Heavy part: stat() MLP (512->1024->1024->1 per pixel, 81920 pixels, twice)
// = ~516 GFLOP -> v_wmma_f32_16x16x32_f16 tensor path, f32 accumulate.
// Everything else is cheap fp32 scalar work.
// ---------------------------------------------------------------------------

typedef _Float16 v8h  __attribute__((ext_vector_type(8)));
typedef _Float16 v16h __attribute__((ext_vector_type(16)));
typedef float    v8f  __attribute__((ext_vector_type(8)));

#define HW   16384              // 128*128
#define IMG  4194304            // 256*HW
#define NPIX 81920              // 5*HW
#define TPX  32                 // pixels per workgroup in stat kernel
#define XS   528                // Xs row stride (halfs), 1056B = 33*32B
#define XA   272                // Xa row stride (halfs), 544B  = 17*32B
#define H1S  1040               // H1 row stride (halfs), 2080B = 65*32B

static constexpr size_t LOSS_OFF = 20971520; // 5*IMG
static constexpr size_t RATE_OFF = 20971521;
static constexpr size_t MASK_OFF = 20971522;

__device__ __forceinline__ float sigmoidf_(float x) { return 1.0f / (1.0f + expf(-x)); }
__device__ __forceinline__ float softplusf_(float x) {
  return fmaxf(x, 0.0f) + log1pf(expf(-fabsf(x)));
}
__device__ __forceinline__ float gauss2(int dy, int dx) {
  // 1/(2*pi*sigma) * exp(-(dx^2+dy^2)/(2*sigma^2)), sigma=1 (NOT normalized)
  return 0.15915494309189535f * expf(-0.5f * (float)(dy * dy + dx * dx));
}

// --- convert stat weights to f16; also zero the mask counter -----------------
__global__ __launch_bounds__(256) void k_wconv(const float* __restrict__ w1,
                                               const float* __restrict__ w2,
                                               _Float16* __restrict__ W1h,
                                               _Float16* __restrict__ W2h,
                                               unsigned* __restrict__ cnt) {
  int idx = blockIdx.x * 256 + threadIdx.x;     // grid covers 1048576
  if (idx == 0) *cnt = 0u;
  if (idx < 524288) W1h[idx] = (_Float16)w1[idx];
  if (idx < 1048576) W2h[idx] = (_Float16)w2[idx];
}

// --- per-(n,c) mean & max over H*W ------------------------------------------
__global__ __launch_bounds__(256) void k_redhw(const float* __restrict__ feat,
                                               float* __restrict__ avg,
                                               float* __restrict__ mxv) {
  __shared__ float ss[256], sm[256];
  int nc = blockIdx.x;                           // 0..1279
  const float* p = feat + (size_t)nc * HW;
  float s = 0.0f, m = -3.4e38f;
  for (int i = threadIdx.x; i < HW; i += 256) { float v = p[i]; s += v; m = fmaxf(m, v); }
  ss[threadIdx.x] = s; sm[threadIdx.x] = m;
  __syncthreads();
  for (int o = 128; o > 0; o >>= 1) {
    if ((int)threadIdx.x < o) {
      ss[threadIdx.x] += ss[threadIdx.x + o];
      sm[threadIdx.x] = fmaxf(sm[threadIdx.x], sm[threadIdx.x + o]);
    }
    __syncthreads();
  }
  if (threadIdx.x == 0) { avg[nc] = ss[0] * (1.0f / HW); mxv[nc] = sm[0]; }
}

// --- channel attention MLP: sigmoid(mlp(avg)+mlp(mx)) -----------------------
__global__ __launch_bounds__(256) void k_chatt(const float* __restrict__ avg,
                                               const float* __restrict__ mxv,
                                               const float* __restrict__ w1,
                                               const float* __restrict__ w2,
                                               float* __restrict__ chatt) {
  __shared__ float hA[80], hM[80];               // 5 images x 16 hidden
  int tid = threadIdx.x;
  if (tid < 160) {
    int n = tid / 32, r = tid % 32;
    int j = r & 15, which = r >> 4;
    const float* in = which ? (mxv + n * 256) : (avg + n * 256);
    const float* wr = w1 + j * 256;              // mlp_w1 (16,256)
    float s = 0.0f;
    for (int c = 0; c < 256; ++c) s += in[c] * wr[c];
    s = s > 0.0f ? s : 0.0f;                     // relu
    if (which) hM[n * 16 + j] = s; else hA[n * 16 + j] = s;
  }
  __syncthreads();
  for (int o = tid; o < 1280; o += 256) {
    int n = o >> 8, c = o & 255;
    const float* wc = w2 + c * 16;               // mlp_w2 (256,16)
    float s = 0.0f;
    for (int j = 0; j < 16; ++j) s += (hA[n * 16 + j] + hM[n * 16 + j]) * wc[j];
    chatt[o] = sigmoidf_(s);
  }
}

// --- per-(n,pixel) channel mean & max ---------------------------------------
__global__ __launch_bounds__(256) void k_redc(const float* __restrict__ feat,
                                              float* __restrict__ cmean,
                                              float* __restrict__ cmax) {
  int gid = blockIdx.x * 256 + threadIdx.x;      // < 81920
  int n = gid >> 14, p = gid & 16383;
  const float* base = feat + (size_t)n * IMG + p;
  float s = 0.0f, m = -3.4e38f;
  for (int c = 0; c < 256; ++c) { float v = base[(size_t)c * HW]; s += v; m = fmaxf(m, v); }
  cmean[gid] = s * (1.0f / 256.0f);
  cmax[gid] = m;
}

// --- sp_att = sigmoid(conv3x3([cmean;cmax], sp_req_w)); act = conv3x3(sigmoid(cmean), g2)
__global__ __launch_bounds__(256) void k_spact(const float* __restrict__ cmean,
                                               const float* __restrict__ cmax,
                                               const float* __restrict__ srw,
                                               float* __restrict__ spatt,
                                               float* __restrict__ act) {
  int gid = blockIdx.x * 256 + threadIdx.x;      // < 81920
  int n = gid >> 14, p = gid & 16383;
  int y = p >> 7, x = p & 127;
  float sa = 0.0f, aa = 0.0f;
  for (int dy = 0; dy < 3; ++dy)
    for (int dx = 0; dx < 3; ++dx) {
      int yy = y + dy - 1, xx = x + dx - 1;
      if (yy < 0 || yy > 127 || xx < 0 || xx > 127) continue;
      int q = n * HW + yy * 128 + xx;
      float cm = cmean[q];
      sa += cm * srw[dy * 3 + dx] + cmax[q] * srw[9 + dy * 3 + dx];
      aa += gauss2(dy - 1, dx - 1) * sigmoidf_(cm);
    }
  spatt[gid] = sigmoidf_(sa);
  act[gid] = aa;
}

// --- ch_coef: sigmoid(ego_ch@Wa.T + ch_att[1:]@Wb.T) then 1D gaussian smooth -
__global__ __launch_bounds__(256) void k_chcoef(const float* __restrict__ chatt,
                                                const float* __restrict__ cf,
                                                float* __restrict__ chcoef) {
  __shared__ float pre[4 * 256];
  int c = threadIdx.x;
  const float* rowA = cf + (size_t)c * 512;      // ch_fus_w (256,512)
  float sA = 0.0f;
  for (int k = 0; k < 256; ++k) sA += (1.0f - chatt[k]) * rowA[k];
  for (int i = 0; i < 4; ++i) {
    float sB = 0.0f;
    const float* att = chatt + (1 + i) * 256;
    for (int k = 0; k < 256; ++k) sB += att[k] * rowA[256 + k];
    pre[i * 256 + c] = sigmoidf_(sA + sB);
  }
  __syncthreads();
  float a = expf(-0.5f);
  float nrm = 1.0f / (1.0f + 2.0f * a);          // normalized 1D gaussian
  float g1a = a * nrm, g1b = nrm;
  for (int i = 0; i < 4; ++i) {
    float L = (c > 0) ? pre[i * 256 + c - 1] : 0.0f;
    float M = pre[i * 256 + c];
    float R = (c < 255) ? pre[i * 256 + c + 1] : 0.0f;
    chcoef[i * 256 + c] = g1a * L + g1b * M + g1a * R;
  }
}

// --- sp_coef = conv3x3(sigmoid(wa*ego_sp + wb*sp_att[1:]), g2) --------------
__global__ __launch_bounds__(256) void k_spcoef(const float* __restrict__ spatt,
                                                const float* __restrict__ spfus,
                                                float* __restrict__ spcoef) {
  int gid = blockIdx.x * 256 + threadIdx.x;      // < 65536
  int i = gid >> 14, p = gid & 16383;
  int y = p >> 7, x = p & 127;
  float wa = spfus[0], wb = spfus[1];
  float acc = 0.0f;
  for (int dy = -1; dy <= 1; ++dy)
    for (int dx = -1; dx <= 1; ++dx) {
      int yy = y + dy, xx = x + dx;
      if (yy < 0 || yy > 127 || xx < 0 || xx > 127) continue;
      int q = yy * 128 + xx;
      float s = sigmoidf_(wa * (1.0f - spatt[q]) + wb * spatt[(1 + i) * HW + q]);
      acc += gauss2(dy, dx) * s;
    }
  spcoef[gid] = acc;
}

// --- mask + sparse_feature + sparse_mask for n=1..4, count mask -------------
__global__ __launch_bounds__(256) void k_mask(const float* __restrict__ feat,
                                              const float* __restrict__ chcoef,
                                              const float* __restrict__ spcoef,
                                              const float* __restrict__ act,
                                              float* __restrict__ out,
                                              unsigned* __restrict__ cnt) {
  __shared__ unsigned sc[256];
  size_t idx = (size_t)blockIdx.x * 256 + threadIdx.x;  // < 16,777,216
  int i = (int)(idx >> 22);
  int rem = (int)(idx & 4194303);
  int c = rem >> 14, p = rem & 16383;
  float sm = chcoef[i * 256 + c] * spcoef[i * HW + p] * act[(1 + i) * HW + p];
  int m = sm > 0.01f ? 1 : 0;
  size_t o = (size_t)(1 + i) * IMG + (size_t)c * HW + p;
  out[o] = m ? feat[o] : 0.0f;
  out[MASK_OFF + o] = m ? 1.0f : 0.0f;
  sc[threadIdx.x] = (unsigned)m;
  __syncthreads();
  for (int s = 128; s > 0; s >>= 1) {
    if ((int)threadIdx.x < s) sc[threadIdx.x] += sc[threadIdx.x + s];
    __syncthreads();
  }
  if (threadIdx.x == 0) atomicAdd(cnt, sc[0]);
}

// --- n=0 plane: sparse_feature[0]=feat[0], sparse_mask[0]=1 -----------------
__global__ __launch_bounds__(256) void k_copy0(const float* __restrict__ feat,
                                               float* __restrict__ out) {
  size_t idx = (size_t)blockIdx.x * 256 + threadIdx.x;  // < 4,194,304
  out[idx] = feat[idx];
  out[MASK_OFF + idx] = 1.0f;
}

// ---------------------------------------------------------------------------
// stat network: per 32-pixel tile, fused 512->1024->1024->1 MLP via WMMA f16.
// Runs both T (feat_n half) and T' (feat_{n+1} half) per tile; sparse half
// of X is shared. Per-WG partial softplus sums -> statP (deterministic sum).
// ---------------------------------------------------------------------------
__global__ __launch_bounds__(256) void k_stat(const float* __restrict__ feat,
                                              const float* __restrict__ sparse,
                                              const _Float16* __restrict__ W1h,
                                              const _Float16* __restrict__ W2h,
                                              const float* __restrict__ b1,
                                              const float* __restrict__ b2,
                                              const float* __restrict__ w3,
                                              const float* __restrict__ b3,
                                              float* __restrict__ statP) {
  __shared__ _Float16 Xs[TPX * XS];   // [pixel][512]  x = [feat_n ; sparse_n]
  __shared__ _Float16 Xa[TPX * XA];   // [pixel][256]  feat_{(n+1)%5}
  __shared__ _Float16 H1[TPX * H1S];  // [pixel][1024] relu(W1 x + b1)
  __shared__ float Tpart[2 * TPX];

  const int tid = threadIdx.x;
  const int wg = blockIdx.x;                // 0..2559
  const int n = wg >> 9;                    // image
  const int p0 = (wg & 511) * TPX;          // pixel base
  const int nnext = (n == 4) ? 0 : n + 1;

  if (tid < 2 * TPX) Tpart[tid] = 0.0f;

  // fill X tiles (fp32 -> f16), global reads coalesced along pixels
  for (int e = tid; e < TPX * 512; e += 256) {
    int c = e >> 5, p = e & 31;
    float v = (c < 256)
        ? feat[(size_t)n * IMG + (size_t)c * HW + p0 + p]
        : sparse[(size_t)n * IMG + (size_t)(c - 256) * HW + p0 + p];
    Xs[p * XS + c] = (_Float16)v;
  }
  for (int e = tid; e < TPX * 256; e += 256) {
    int c = e >> 5, p = e & 31;
    Xa[p * XA + c] = (_Float16)feat[(size_t)nnext * IMG + (size_t)c * HW + p0 + p];
  }
  __syncthreads();

  const int lane = tid & 31;
  const int wave = tid >> 5;
  const int row = lane & 15;                 // A: M row
  const int col = lane & 15;                 // B/C/D: N column (pixel)
  const int kbA = (lane < 16) ? 0 : 16;      // A frag: K 0-15 vs 16-31
  const int kbB = (lane < 16) ? 0 : 16;      // B frag: K 0-15 vs 16-31
  const int rbase = (lane < 16) ? 0 : 8;     // C/D frag: M 0-7 vs 8-15

  for (int run = 0; run < 2; ++run) {
    // ---- layer 1: H1 = relu(W1 X + b1), 64 m-tiles x 2 n-tiles ----
    for (int tt = wave; tt < 128; tt += 8) {
      int m0 = (tt >> 1) * 16;
      int n0 = (tt & 1) * 16;
      v8f acc = {0.f, 0.f, 0.f, 0.f, 0.f, 0.f, 0.f, 0.f};
      for (int ks = 0; ks < 16; ++ks) {
        int k0 = ks * 32;
        const _Float16* ap = W1h + (size_t)(m0 + row) * 512 + (k0 + kbA);
        union { v16h v; struct { v8h lo, hi; } s; } A;
        A.s.lo = *(const v8h*)(ap);
        A.s.hi = *(const v8h*)(ap + 8);
        const _Float16* bp = (run == 0 || k0 >= 256)
            ? &Xs[(n0 + col) * XS + k0 + kbB]
            : &Xa[(n0 + col) * XA + k0 + kbB];
        v16h B = *(const v16h*)bp;
        acc = __builtin_amdgcn_wmma_f32_16x16x32_f16(false, A.v, false, B,
                                                     (short)0, acc, false, false);
      }
      v8h h;
#pragma unroll
      for (int r = 0; r < 8; ++r) {
        float v = acc[r] + b1[m0 + rbase + r];
        h[r] = (_Float16)(v > 0.0f ? v : 0.0f);
      }
      *(v8h*)&H1[(n0 + col) * H1S + m0 + rbase] = h;
    }
    __syncthreads();

    // ---- layer 2 (+ layer 3 folded into epilogue) ----
    for (int tt = wave; tt < 128; tt += 8) {
      int m0 = (tt >> 1) * 16;
      int n0 = (tt & 1) * 16;
      v8f acc = {0.f, 0.f, 0.f, 0.f, 0.f, 0.f, 0.f, 0.f};
      for (int ks = 0; ks < 32; ++ks) {
        int k0 = ks * 32;
        const _Float16* ap = W2h + (size_t)(m0 + row) * 1024 + (k0 + kbA);
        union { v16h v; struct { v8h lo, hi; } s; } A;
        A.s.lo = *(const v8h*)(ap);
        A.s.hi = *(const v8h*)(ap + 8);
        v16h B = *(const v16h*)&H1[(n0 + col) * H1S + k0 + kbB];
        acc = __builtin_amdgcn_wmma_f32_16x16x32_f16(false, A.v, false, B,
                                                     (short)0, acc, false, false);
      }
      float part = 0.0f;
#pragma unroll
      for (int r = 0; r < 8; ++r) {
        int oc = m0 + rbase + r;
        float v = acc[r] + b2[oc];
        v = v > 0.0f ? v : 0.0f;
        part += v * w3[oc];
      }
      atomicAdd(&Tpart[run * TPX + n0 + col], part);
    }
    __syncthreads();   // also protects H1 before next run overwrites it
  }

  // ---- softplus + per-WG partial sums ----
  float s1 = 0.0f, s2 = 0.0f;
  if (tid < TPX) {
    float b3v = b3[0];
    float T = Tpart[tid] + b3v;          // run 0:  [feat ; sparse]
    float Tp = Tpart[TPX + tid] + b3v;   // run 1:  [feat' ; sparse]
    s1 = softplusf_(-T);                 // -> mean(softplus(-T))
    s2 = softplusf_(Tp);                 // -> mean(softplus(T'))
  }
  __syncthreads();
  if (tid < TPX) { Tpart[tid] = s1; Tpart[TPX + tid] = s2; }
  __syncthreads();
  if (tid == 0) {
    float a = 0.0f, b = 0.0f;
    for (int i = 0; i < TPX; ++i) { a += Tpart[i]; b += Tpart[TPX + i]; }
    statP[wg] = a;
    statP[2560 + wg] = b;
  }
}

// --- finalize scalars -------------------------------------------------------
__global__ void k_final(const float* __restrict__ statP,
                        const unsigned* __restrict__ cnt,
                        float* __restrict__ out) {
  if (threadIdx.x == 0 && blockIdx.x == 0) {
    float s1 = 0.0f, s2 = 0.0f;
    for (int i = 0; i < 2560; ++i) { s1 += statP[i]; s2 += statP[2560 + i]; }
    // total_loss = mean(softplus(-T)) + mean(softplus(T'))
    out[LOSS_OFF] = (s1 + s2) * (1.0f / 81920.0f);
    out[RATE_OFF] = (float)(*cnt) * (1.0f / 16777216.0f);
  }
}

extern "C" void kernel_launch(void* const* d_in, const int* in_sizes, int n_in,
                              void* d_out, int out_size, void* d_ws, size_t ws_size,
                              hipStream_t stream) {
  (void)in_sizes; (void)n_in; (void)out_size; (void)ws_size;
  const float* feat    = (const float*)d_in[0];
  const float* mlp_w1  = (const float*)d_in[1];
  const float* mlp_w2  = (const float*)d_in[2];
  const float* sp_req  = (const float*)d_in[3];
  const float* ch_fus  = (const float*)d_in[4];
  const float* sp_fus  = (const float*)d_in[5];
  const float* st_w1   = (const float*)d_in[6];
  const float* st_b1   = (const float*)d_in[7];
  const float* st_w2   = (const float*)d_in[8];
  const float* st_b2   = (const float*)d_in[9];
  const float* st_w3   = (const float*)d_in[10];
  const float* st_b3   = (const float*)d_in[11];
  float* out = (float*)d_out;

  // workspace layout
  _Float16* W1h = (_Float16*)d_ws;               // 1024*512
  _Float16* W2h = W1h + 1024 * 512;              // 1024*1024
  float* wsF = (float*)(W2h + 1024 * 1024);
  float* avg    = wsF;                // 1280
  float* mxv    = avg + 1280;         // 1280
  float* chatt  = mxv + 1280;         // 1280
  float* cmean  = chatt + 1280;       // 81920
  float* cmax   = cmean + 81920;      // 81920
  float* spatt  = cmax + 81920;       // 81920
  float* act    = spatt + 81920;      // 81920
  float* chcoef = act + 81920;        // 1024
  float* spcoef = chcoef + 1024;      // 65536
  float* statP  = spcoef + 65536;     // 5120
  unsigned* cnt = (unsigned*)(statP + 5120);

  k_wconv <<<4096, 256, 0, stream>>>(st_w1, st_w2, W1h, W2h, cnt);
  k_redhw <<<1280, 256, 0, stream>>>(feat, avg, mxv);
  k_chatt <<<1, 256, 0, stream>>>(avg, mxv, mlp_w1, mlp_w2, chatt);
  k_redc  <<<320, 256, 0, stream>>>(feat, cmean, cmax);
  k_spact <<<320, 256, 0, stream>>>(cmean, cmax, sp_req, spatt, act);
  k_chcoef<<<1, 256, 0, stream>>>(chatt, ch_fus, chcoef);
  k_spcoef<<<256, 256, 0, stream>>>(spatt, sp_fus, spcoef);
  k_mask  <<<65536, 256, 0, stream>>>(feat, chcoef, spcoef, act, out, cnt);
  k_copy0 <<<16384, 256, 0, stream>>>(feat, out);
  k_stat  <<<2560, 256, 0, stream>>>(feat, out, W1h, W2h, st_b1, st_b2, st_w3,
                                     st_b3, statP);
  k_final <<<1, 1, 0, stream>>>(statP, cnt, out);
}
